// Net_75230647156978
// MI455X (gfx1250) — matrix operand order repaired and checked by reading, not compiled, not run on hardware
//
#include <hip/hip_runtime.h>
#include <hip/hip_bf16.h>

#define BATCH_TILE 16
#define XDIM 64
#define HID 24
#define GH 8

typedef _Float16 v16h __attribute__((ext_vector_type(16)));
typedef _Float16 v8h  __attribute__((ext_vector_type(8)));
typedef float    v8f  __attribute__((ext_vector_type(8)));
typedef float    v4f  __attribute__((ext_vector_type(4)));

union FragU { v16h v; v8h h[2]; };

__device__ __forceinline__ float fast_tanh(float x) {
#if defined(__AMDGCN__) && __has_builtin(__builtin_amdgcn_tanhf)
  return __builtin_amdgcn_tanhf(x);        // v_tanh_f32 on gfx1250
#else
  const float e = __expf(2.0f * x);
  return 1.0f - 2.0f / (e + 1.0f);
#endif
}

__device__ __forceinline__ v8f wmma_f16(v16h a, v16h b, v8f c) {
  // D = A(16x32 f16) * B(32x16 f16) + C(16x16 f32)
  return __builtin_amdgcn_wmma_f32_16x16x32_f16(
      false, a, false, b, (short)0, c, false, false);
}

__device__ __forceinline__ v8f cinit(float s) {
  v8f c;
#pragma unroll
  for (int v = 0; v < 8; ++v) c[v] = s;
  return c;
}

__device__ __forceinline__ v8f vtanh(v8f a) {
  v8f r;
#pragma unroll
  for (int v = 0; v < 8; ++v) r[v] = fast_tanh(a[v]);
  return r;
}

// Low 32 bits of a generic pointer into LDS == hardware LDS byte offset
__device__ __forceinline__ uint32_t lds_off32(const void* p) {
  return (uint32_t)(uintptr_t)p;
}

// Async-copy one 16x64 f32 tile (4KB) global -> LDS, fully coalesced:
// 32 lanes x 16B = 512B per instruction, 8 instructions; the immediate
// offset is added to BOTH the LDS and global addresses.
__device__ __forceinline__ void async_stage_x(uint32_t lds_base_off,
                                              const float* __restrict__ gsrc,
                                              int lane) {
  const uint32_t ldsoff = lds_base_off + (uint32_t)(lane * 16);
  const char* g = (const char*)gsrc + lane * 16;
  asm volatile(
      "global_load_async_to_lds_b128 %0, %1, off\n\t"
      "global_load_async_to_lds_b128 %0, %1, off offset:512\n\t"
      "global_load_async_to_lds_b128 %0, %1, off offset:1024\n\t"
      "global_load_async_to_lds_b128 %0, %1, off offset:1536\n\t"
      "global_load_async_to_lds_b128 %0, %1, off offset:2048\n\t"
      "global_load_async_to_lds_b128 %0, %1, off offset:2560\n\t"
      "global_load_async_to_lds_b128 %0, %1, off offset:3072\n\t"
      "global_load_async_to_lds_b128 %0, %1, off offset:3584"
      :: "v"(ldsoff), "v"(g) : "memory");
}

// B-fragment (32x16 tile of W[K x N], row-major), zero-padded.
// Lane holds column n = NB + (lane&15); lanes 0-15 hold K = KB+0..15,
// lanes 16-31 hold K = KB+16..31, packed 2 per VGPR in ascending K.
__device__ __forceinline__ v16h load_bfrag(const float* __restrict__ W,
                                           int Kact, int Nact,
                                           int KB, int NB, int lane) {
  const int n  = NB + (lane & 15);
  const int kb = KB + ((lane & 16) ? 16 : 0);
  v16h f;
#pragma unroll
  for (int i = 0; i < 16; ++i) {
    const int k = kb + i;
    float w = ((k < Kact) && (n < Nact)) ? W[k * Nact + n] : 0.0f;
    f[i] = (_Float16)w;
  }
  return f;
}

// A-fragment (16x32 f16) from the f32 LDS staging tile [16][64].
// Lane row r = lane&15; lanes 0-15: K = KB+0..15;
// lanes 16-31: K = KB+16..31.
__device__ __forceinline__ v16h load_afrag_stage(const float* stage,
                                                 int KB, int lane) {
  const int r    = lane & 15;
  const int koff = (lane & 16) ? 16 : 0;
  const float* p = stage + r * XDIM + KB + koff;
  const v4f a0 = *(const v4f*)(p + 0);
  const v4f a1 = *(const v4f*)(p + 4);
  const v4f b0 = *(const v4f*)(p + 8);
  const v4f b1 = *(const v4f*)(p + 12);
  v16h f;
#pragma unroll
  for (int i = 0; i < 4; ++i) {
    f[i]      = (_Float16)a0[i];
    f[4 + i]  = (_Float16)a1[i];
    f[8 + i]  = (_Float16)b0[i];
    f[12 + i] = (_Float16)b1[i];
  }
  return f;
}

// A-fragment from a wave-private LDS tile [16 rows][32 cols] of f16.
__device__ __forceinline__ v16h load_afrag_lds(const _Float16* tile, int lane) {
  const int r    = lane & 15;
  const int koff = (lane & 16) ? 16 : 0;
  const _Float16* p = tile + r * 32 + koff;
  FragU u;
  u.h[0] = *(const v8h*)(p);
  u.h[1] = *(const v8h*)(p + 8);
  return u.v;
}

// Pre-built weight fragment stored in LDS as [32 lanes][16 halves].
__device__ __forceinline__ v16h load_wfrag_lds(const _Float16* wf, int lane) {
  const _Float16* p = wf + lane * 16;
  FragU u;
  u.h[0] = *(const v8h*)(p);
  u.h[1] = *(const v8h*)(p + 8);
  return u.v;
}

__device__ __forceinline__ void store_wfrag_lds(_Float16* wf, v16h f, int lane) {
  FragU u; u.v = f;
  *(v8h*)(wf + lane * 16)     = u.h[0];
  *(v8h*)(wf + lane * 16 + 8) = u.h[1];
}

// D-fragment (16x16 f32) -> LDS tile as f16; pads columns >= Nact with zero.
// D layout: lanes 0-15: N = NB+lane, M = v; lanes 16-31: N = NB+lane-16, M = v+8.
__device__ __forceinline__ void store_dfrag_lds(_Float16* tile, v8f d,
                                                int NB, int Nact, int lane) {
  const int n  = NB + (lane & 15);
  const int mb = (lane & 16) ? 8 : 0;
  const bool ok = (n < Nact);
#pragma unroll
  for (int v = 0; v < 8; ++v) {
    tile[(mb + v) * 32 + n] = (_Float16)(ok ? d[v] : 0.0f);
  }
}

__device__ __forceinline__ float bias_lane(const float* __restrict__ b,
                                           int Nact, int NB, int lane) {
  const int n = NB + (lane & 15);
  return (n < Nact) ? b[n] : 0.0f;
}

__global__ __launch_bounds__(256)
void moe_forward_kernel(const float* __restrict__ x,
                        const float* __restrict__ bbW1, const float* __restrict__ bbb1,
                        const float* __restrict__ bbW2, const float* __restrict__ bbb2,
                        const float* __restrict__ e1W,  const float* __restrict__ e1b,
                        const float* __restrict__ e2W,  const float* __restrict__ e2b,
                        const float* __restrict__ gW1,  const float* __restrict__ gb1,
                        const float* __restrict__ gW2,  const float* __restrict__ gb2,
                        const float* __restrict__ hsW,  const float* __restrict__ hsb,
                        const float* __restrict__ htW,  const float* __restrict__ htb,
                        const float* __restrict__ hyW,  const float* __restrict__ hyb,
                        const float* __restrict__ base,
                        float* __restrict__ out, int ntiles) {
  // Per-wave LDS: double-buffered x staging (2x4KB) + h tile + gate tile.
  __shared__ __attribute__((aligned(16))) float    lds_stage[8][2][BATCH_TILE * XDIM];
  __shared__ __attribute__((aligned(16))) _Float16 lds_h[8][16 * 32];
  __shared__ __attribute__((aligned(16))) _Float16 lds_g[8][16 * 32];
  // Block-shared cold weight fragments: 0=fG1_0, 1=fG1_1, 2=fG2, 3=fH.
  __shared__ __attribute__((aligned(16))) _Float16 lds_wf[4][32 * 16];

  const int lane = threadIdx.x & 31;
  const int wid  = threadIdx.x >> 5;
  _Float16* myh = &lds_h[wid][0];
  _Float16* myg = &lds_g[wid][0];
  const uint32_t stg0_off = lds_off32(&lds_stage[wid][0][0]);

  // Zero the gate tile once; columns 16..31 stay zero (K padding 8->32).
  {
    int* g32 = (int*)myg;
#pragma unroll
    for (int i = 0; i < 8; ++i) g32[lane + 32 * i] = 0;
  }

  // Wave 0 builds the cold fragments into LDS (gate + merged head matrix).
  if (wid == 0) {
    store_wfrag_lds(&lds_wf[0][0], load_bfrag(gW1, XDIM, GH, 0, 0, lane), lane);
    store_wfrag_lds(&lds_wf[1][0], load_bfrag(gW1, XDIM, GH, 32, 0, lane), lane);
    store_wfrag_lds(&lds_wf[2][0], load_bfrag(gW2, GH, 2, 0, 0, lane), lane);
    v16h fH;  // merged [24 x 3] head matrix (strain, tensile, yield), padded
    {
      const int n  = lane & 15;
      const int kb = (lane & 16) ? 16 : 0;
#pragma unroll
      for (int i = 0; i < 16; ++i) {
        const int k = kb + i;
        float w = 0.0f;
        if (k < HID) {
          if (n == 0) w = hsW[k];
          else if (n == 1) w = htW[k];
          else if (n == 2) w = hyW[k];
        }
        fH[i] = (_Float16)w;
      }
    }
    store_wfrag_lds(&lds_wf[3][0], fH, lane);
  }

  // ---- hot weight fragments resident in VGPRs (10 x 8 = 80 VGPRs) ----
  const v16h fW1_00 = load_bfrag(bbW1, XDIM, HID, 0, 0, lane);
  const v16h fW1_10 = load_bfrag(bbW1, XDIM, HID, 32, 0, lane);
  const v16h fW1_01 = load_bfrag(bbW1, XDIM, HID, 0, 16, lane);
  const v16h fW1_11 = load_bfrag(bbW1, XDIM, HID, 32, 16, lane);
  const v16h fW2_0  = load_bfrag(bbW2, HID, HID, 0, 0, lane);
  const v16h fW2_1  = load_bfrag(bbW2, HID, HID, 0, 16, lane);
  const v16h fE1_0  = load_bfrag(e1W, HID, HID, 0, 0, lane);
  const v16h fE1_1  = load_bfrag(e1W, HID, HID, 0, 16, lane);
  const v16h fE2_0  = load_bfrag(e2W, HID, HID, 0, 0, lane);
  const v16h fE2_1  = load_bfrag(e2W, HID, HID, 0, 16, lane);

  const float b_bb1_0 = bias_lane(bbb1, HID, 0, lane);
  const float b_bb1_1 = bias_lane(bbb1, HID, 16, lane);
  const float b_bb2_0 = bias_lane(bbb2, HID, 0, lane);
  const float b_bb2_1 = bias_lane(bbb2, HID, 16, lane);
  const float b_e1_0  = bias_lane(e1b, HID, 0, lane);
  const float b_e1_1  = bias_lane(e1b, HID, 16, lane);
  const float b_e2_0  = bias_lane(e2b, HID, 0, lane);
  const float b_e2_1  = bias_lane(e2b, HID, 16, lane);
  const float b_g1    = bias_lane(gb1, GH, 0, lane);
  const float b_g2    = bias_lane(gb2, 2, 0, lane);
  float b_head;
  {
    const int n = lane & 15;
    b_head = (n == 0) ? (hsb[0] + base[0])
           : (n == 1) ? (htb[0] + base[1])
           : (n == 2) ? (hyb[0] + base[2]) : 0.0f;
  }

  __syncthreads();  // cold fragments visible to all waves

  const int gwave = blockIdx.x * (blockDim.x >> 5) + wid;
  const int nw    = gridDim.x * (blockDim.x >> 5);

  // Prologue: start async copy of first tile into buffer 0.
  if (gwave < ntiles)
    async_stage_x(stg0_off, x + (size_t)gwave * BATCH_TILE * XDIM, lane);

  int cur = 0;
  for (int t = gwave; t < ntiles; t += nw) {
    const int row0 = t * BATCH_TILE;
    const int tn   = t + nw;

    if (tn < ntiles) {
      // Kick off next tile's copy into the other buffer, then release the
      // current buffer (async loads complete in order -> cnt<=8 is ours).
      async_stage_x(stg0_off + (uint32_t)((cur ^ 1) << 12),
                    x + (size_t)tn * BATCH_TILE * XDIM, lane);
      asm volatile("s_wait_asynccnt 0x8" ::: "memory");
    } else {
      asm volatile("s_wait_asynccnt 0x0" ::: "memory");
    }
    const float* st = lds_stage[wid][cur];  // direct shared indexing -> ds loads

    const v16h ax0 = load_afrag_stage(st, 0, lane);
    const v16h ax1 = load_afrag_stage(st, 32, lane);

    // ---- gate first (so expert mixing needs no long-lived fragments) ----
    v8f g = cinit(b_g1);
    g = wmma_f16(ax0, load_wfrag_lds(&lds_wf[0][0], lane), g);
    g = wmma_f16(ax1, load_wfrag_lds(&lds_wf[1][0], lane), g);
    store_dfrag_lds(myg, vtanh(g), 0, GH, lane);
    const v16h agf = load_afrag_lds(myg, lane);
    v8f gl = cinit(b_g2);
    gl = wmma_f16(agf, load_wfrag_lds(&lds_wf[2][0], lane), gl);

    // 2-way softmax: w1 = sigmoid(l1 - l0); logits sit in cols 0/1 of each half
    const int hb = lane & 16;
    v8f W1v;
#pragma unroll
    for (int v = 0; v < 8; ++v) {
      const float l0 = __shfl(gl[v], hb | 0, 32);
      const float l1 = __shfl(gl[v], hb | 1, 32);
      W1v[v] = 1.0f / (1.0f + __expf(l0 - l1));
    }

    // ---- backbone layer 1: [16x64] @ [64x24] + b, tanh ----
    v8f h0 = cinit(b_bb1_0);
    h0 = wmma_f16(ax0, fW1_00, h0);
    h0 = wmma_f16(ax1, fW1_10, h0);
    v8f h1 = cinit(b_bb1_1);
    h1 = wmma_f16(ax0, fW1_01, h1);
    h1 = wmma_f16(ax1, fW1_11, h1);
    store_dfrag_lds(myh, vtanh(h0), 0, HID, lane);
    store_dfrag_lds(myh, vtanh(h1), 16, HID, lane);
    const v16h ah1 = load_afrag_lds(myh, lane);

    // ---- backbone layer 2: [16x24] @ [24x24] + b, tanh ----
    v8f k0 = cinit(b_bb2_0); k0 = wmma_f16(ah1, fW2_0, k0);
    v8f k1 = cinit(b_bb2_1); k1 = wmma_f16(ah1, fW2_1, k1);
    store_dfrag_lds(myh, vtanh(k0), 0, HID, lane);
    store_dfrag_lds(myh, vtanh(k1), 16, HID, lane);
    const v16h ah = load_afrag_lds(myh, lane);

    // ---- experts + immediate mix, one N-tile at a time ----
    {
      v8f e1 = cinit(b_e1_0); e1 = wmma_f16(ah, fE1_0, e1); e1 = vtanh(e1);
      v8f e2 = cinit(b_e2_0); e2 = wmma_f16(ah, fE2_0, e2); e2 = vtanh(e2);
      v8f m;
#pragma unroll
      for (int v = 0; v < 8; ++v) m[v] = e1[v] + W1v[v] * (e2[v] - e1[v]);
      store_dfrag_lds(myh, m, 0, HID, lane);
    }
    {
      v8f e1 = cinit(b_e1_1); e1 = wmma_f16(ah, fE1_1, e1); e1 = vtanh(e1);
      v8f e2 = cinit(b_e2_1); e2 = wmma_f16(ah, fE2_1, e2); e2 = vtanh(e2);
      v8f m;
#pragma unroll
      for (int v = 0; v < 8; ++v) m[v] = e1[v] + W1v[v] * (e2[v] - e1[v]);
      store_dfrag_lds(myh, m, 16, HID, lane);
    }
    const v16h am = load_afrag_lds(myh, lane);

    // ---- heads: [16x24] @ [24x3] (+ bias + base, folded into C) ----
    v8f od = cinit(b_head);
    od = wmma_f16(am, load_wfrag_lds(&lds_wf[3][0], lane), od);

    const int n = lane & 15;
    if (n < 3) {
      const int mb = hb ? 8 : 0;
#pragma unroll
      for (int v = 0; v < 8; ++v) {
        out[(size_t)(row0 + mb + v) * 3 + n] = od[v];
      }
    }

    cur ^= 1;
  }
}

extern "C" void kernel_launch(void* const* d_in, const int* in_sizes, int n_in,
                              void* d_out, int out_size, void* d_ws, size_t ws_size,
                              hipStream_t stream) {
  const float* x    = (const float*)d_in[0];
  const float* bbW1 = (const float*)d_in[1];
  const float* bbb1 = (const float*)d_in[2];
  const float* bbW2 = (const float*)d_in[3];
  const float* bbb2 = (const float*)d_in[4];
  const float* e1W  = (const float*)d_in[5];
  const float* e1b  = (const float*)d_in[6];
  const float* e2W  = (const float*)d_in[7];
  const float* e2b  = (const float*)d_in[8];
  const float* gW1  = (const float*)d_in[9];
  const float* gb1  = (const float*)d_in[10];
  const float* gW2  = (const float*)d_in[11];
  const float* gb2  = (const float*)d_in[12];
  const float* hsW  = (const float*)d_in[13];
  const float* hsb  = (const float*)d_in[14];
  const float* htW  = (const float*)d_in[15];
  const float* htb  = (const float*)d_in[16];
  const float* hyW  = (const float*)d_in[17];
  const float* hyb  = (const float*)d_in[18];
  const float* base = (const float*)d_in[19];
  float* out = (float*)d_out;

  const int Bn     = in_sizes[0] / XDIM;
  const int ntiles = Bn / BATCH_TILE;

  int blocks = (ntiles + 7) / 8;   // 8 waves per block -> ~1 tile per wave
  if (blocks > 1024) blocks = 1024;
  if (blocks < 1) blocks = 1;

  moe_forward_kernel<<<blocks, 256, 0, stream>>>(
      x, bbW1, bbb1, bbW2, bbb2, e1W, e1b, e2W, e2b,
      gW1, gb1, gW2, gb2, hsW, hsb, htW, htb, hyW, hyb, base,
      out, ntiles);
}
